// DiscriminatorBlock_60155311947963
// MI455X (gfx1250) — compile-verified, matrix operand order checked by reading, not run on hardware
//
#include <hip/hip_runtime.h>

typedef __attribute__((ext_vector_type(16))) __bf16          v16bf;
typedef __attribute__((ext_vector_type(8)))  float           v8f;
typedef __attribute__((ext_vector_type(4)))  float           f32x4;
typedef __attribute__((ext_vector_type(8)))  unsigned short  u16x8;
typedef __attribute__((ext_vector_type(16))) unsigned short  u16x16;
typedef __attribute__((ext_vector_type(4)))  unsigned int    u32x4;
typedef __attribute__((ext_vector_type(8)))  int             i32x8;
typedef __attribute__((ext_vector_type(4)))  int             i32x4;
typedef unsigned short u16;

#define NB 8
#define HH 512
#define WW 512
#define CI 32
#define CO 32
#define HO 256
#define WO 256

#if defined(__has_builtin)
#if __has_builtin(__builtin_amdgcn_tensor_load_to_lds) && \
    __has_builtin(__builtin_amdgcn_s_wait_tensorcnt)
#define USE_TDM 1
#endif
#endif

__device__ __forceinline__ u16 f2bf(float f) {
  unsigned u = __builtin_bit_cast(unsigned, f);
  u += 0x7FFFu + ((u >> 16) & 1u);          // round to nearest even
  return (u16)(u >> 16);
}

__device__ __forceinline__ float lrelu_gain(float v) {
  return (v < 0.0f ? 0.2f * v : v) * 1.4142135623730951f;  // leaky_relu * sqrt(2)
}

__device__ __forceinline__ v16bf cat16(u16x8 lo, u16x8 hi) {
  return __builtin_bit_cast(v16bf,
      __builtin_shufflevector(lo, hi, 0,1,2,3,4,5,6,7,8,9,10,11,12,13,14,15));
}

// ---------------------------------------------------------------------------
// Kernel 0: fold eq-lr scale + FIR into bf16 weight tables, stored [tap][n][k]
// so B-fragments are contiguous 16B runs per lane.
// ---------------------------------------------------------------------------
__global__ void prep_weights(const float* __restrict__ w0,
                             const float* __restrict__ w1,
                             const float* __restrict__ wres,
                             u16* __restrict__ w0b,    // [9][32 n][32 k]
                             u16* __restrict__ weff,   // [36][32 n][32 k]
                             u16* __restrict__ wresb)  // [32 n][32 k]
{
  const float EQ  = 0.058925565098878960f;   // 1/sqrt(3*3*32)
  const float EQR = 0.176776695296636880f;   // 1/sqrt(32)
  const float K4[4] = {0.125f, 0.375f, 0.375f, 0.125f};  // normalized FIR taps
  int tid = blockIdx.x * blockDim.x + threadIdx.x;
  int stride = gridDim.x * blockDim.x;

  for (int idx = tid; idx < 9 * 32 * 32; idx += stride) {
    int t = idx >> 10, n = (idx >> 5) & 31, k = idx & 31;
    w0b[idx] = f2bf(w0[(t * 32 + k) * 32 + n] * EQ);     // HWIO -> [t][n][k]
  }
  for (int idx = tid; idx < 36 * 32 * 32; idx += stride) {
    int t = idx >> 10, n = (idx >> 5) & 31, k = idx & 31;
    int u = t / 6, v = t % 6;
    float s = 0.0f;
    for (int dy = 0; dy < 3; ++dy) {
      int i = u - dy; if (i < 0 || i > 3) continue;
      for (int dx = 0; dx < 3; ++dx) {
        int j = v - dx; if (j < 0 || j > 3) continue;
        s += w1[((dy * 3 + dx) * 32 + k) * 32 + n] * (K4[i] * K4[j]);
      }
    }
    weff[idx] = f2bf(s * EQ);
  }
  for (int idx = tid; idx < 32 * 32; idx += stride) {
    int n = idx >> 5, k = idx & 31;
    wresb[idx] = f2bf(wres[k * 32 + n] * EQR);
  }
}

// ---------------------------------------------------------------------------
// Kernel 1: 3x3 conv stride1 SAME + bias + lrelu*sqrt2, output h in bf16.
// Workgroup = 256 threads = 8 waves, covers 128 pixels of one row.
// Implicit GEMM: each conv tap = one 16x16x32 bf16 WMMA K-slice (x2 N-tiles).
// ---------------------------------------------------------------------------
__global__ void conv0_lrelu(const float* __restrict__ x,
                            const float* __restrict__ bias,
                            const u16* __restrict__ w0b,
                            u16* __restrict__ h)
{
  extern __shared__ u16 smem[];              // [3 rows][130 cols][32 ch] bf16
  const int wg  = blockIdx.x;
  const int xb0 = (wg & 3) * 128;
  const int row = (wg >> 2) & 511;
  const int n   = wg >> 11;
  const int tid = threadIdx.x;

  // Stage x halo tile into LDS as bf16 (zero-pad outside the image).
  for (int u = tid; u < 3 * 130 * 4; u += 256) {
    int ck = u & 3;
    int c  = (u >> 2) % 130;
    int r  = u / (130 * 4);
    int gy = row - 1 + r;
    int gx = xb0 - 1 + c;
    u16x8 pk;
    if (gy >= 0 && gy < HH && gx >= 0 && gx < WW) {
      const float* src = x + (((size_t)(n * HH + gy) * WW + gx) * CI + ck * 8);
      f32x4 f0 = *(const f32x4*)src;
      f32x4 f1 = *(const f32x4*)(src + 4);
      pk = (u16x8){f2bf(f0.x), f2bf(f0.y), f2bf(f0.z), f2bf(f0.w),
                   f2bf(f1.x), f2bf(f1.y), f2bf(f1.z), f2bf(f1.w)};
    } else {
      pk = (u16x8)0;
    }
    *(u16x8*)(smem + ((r * 130 + c) * 32 + ck * 8)) = pk;
  }
  __syncthreads();

  const int wave = tid >> 5;
  const int lane = tid & 31;
  const int m    = lane & 15;                // A row / D column
  const int hi   = lane >> 4;
  const int run0 = hi * 8;                   // K = hi*8 + 0..7
  const int run1 = 16 + hi * 8;              // K = 16 + hi*8 + 0..7

  v8f acc0 = {0,0,0,0,0,0,0,0};
  v8f acc1 = {0,0,0,0,0,0,0,0};

  #pragma unroll
  for (int t = 0; t < 9; ++t) {
    const int dy = t / 3, dx = t % 3;
    const u16* ap = smem + ((dy * 130 + (wave * 16 + m + dx)) * 32);
    v16bf a = cat16(*(const u16x8*)(ap + run0), *(const u16x8*)(ap + run1));

    const u16* bp = w0b + (t * 1024 + m * 32);
    v16bf bv0 = cat16(*(const u16x8*)(bp + run0), *(const u16x8*)(bp + run1));
    v16bf bv1 = cat16(*(const u16x8*)(bp + 512 + run0),
                      *(const u16x8*)(bp + 512 + run1));

    acc0 = __builtin_amdgcn_wmma_f32_16x16x32_bf16(false, a, false, bv0,
                                                   (short)0, acc0, false, false);
    acc1 = __builtin_amdgcn_wmma_f32_16x16x32_bf16(false, a, false, bv1,
                                                   (short)0, acc1, false, false);
  }

  const float bias0 = bias[m];
  const float bias1 = bias[m + 16];
  u16* hp = h + (((size_t)(n * HH + row) * WW + (xb0 + wave * 16 + hi * 8)) * CO + m);
  #pragma unroll
  for (int g = 0; g < 8; ++g) {
    hp[(size_t)g * CO]      = f2bf(lrelu_gain(acc0[g] + bias0));
    hp[(size_t)g * CO + 16] = f2bf(lrelu_gain(acc1[g] + bias1));
  }
}

// ---------------------------------------------------------------------------
// Kernel 2: fused blur+3x3 stride2 (== 6x6 stride2 with w_eff) + bias + lrelu,
// plus residual (4x4 FIR gather + 1x1 GEMM), merged with sqrt(0.5).
// Workgroup = 8 waves, covers 128 output pixels of one output row.
// Interior tiles stage h via the Tensor Data Mover (bf16 copy, no conversion);
// border tiles (need zero-padding) use the manual path.
// ---------------------------------------------------------------------------
__global__ void down_conv_res(const u16* __restrict__ h,
                              const float* __restrict__ x,
                              const float* __restrict__ bias,
                              const u16* __restrict__ weff,
                              const u16* __restrict__ wresb,
                              float* __restrict__ out)
{
  extern __shared__ u16 smem[];              // [6 rows][260 cols][32 ch] bf16
  const int wg  = blockIdx.x;
  const int ob0 = (wg & 1) * 128;
  const int oy  = (wg >> 1) & 255;
  const int n   = wg >> 9;
  const int tid = threadIdx.x;

  const int gy0 = 2 * oy - 2;                // first h row of the tile
  const int gx0 = 2 * ob0 - 2;               // first h col of the tile
  const bool interior =
      (gy0 >= 0) && (gy0 + 6 <= HH) && (gx0 >= 0) && (gx0 + 260 <= WW);

#ifdef USE_TDM
  if (interior) {
    if (tid < 32) {                          // one TDM issue from wave 0
      // ---- D# group 0: count=1, lds_addr, global_addr, type=2 ("image")
      unsigned long long ga =
          (unsigned long long)(size_t)(h + (((size_t)(n * HH + gy0) * WW + gx0) * CO));
      unsigned lds_off = (unsigned)(size_t)smem;   // LDS aperture: low 32 bits
      u32x4 g0;
      g0[0] = 1u;                                  // count = 1 valid descriptor
      g0[1] = lds_off;                             // lds_addr (bytes)
      g0[2] = (unsigned)(ga & 0xFFFFFFFFu);        // global_addr[31:0]
      g0[3] = (unsigned)((ga >> 32) & 0x1FFFFFFu)  // global_addr[56:32]
              | (2u << 30);                        // type = 2
      // ---- D# group 1: 2D tile, data_size = 2 bytes
      const unsigned td0 = WW * CO;                // tensor_dim0 = 16384 elems
      const unsigned td1 = HH;                     // tensor_dim1 = 512 rows
      const unsigned tl0 = 260 * CO;               // tile_dim0  = 8320 elems
      const unsigned tl1 = 6;                      // tile_dim1  = 6 rows
      const unsigned st0 = WW * CO;                // tensor_dim0_stride
      i32x8 g1;
      g1[0] = (int)(1u << 16);                     // data_size = 1 -> 2 bytes
      g1[1] = (int)((td0 & 0xFFFFu) << 16);        // tensor_dim0[15:0]
      g1[2] = (int)((td0 >> 16) | ((td1 & 0xFFFFu) << 16));
      g1[3] = (int)((td1 >> 16) | (tl0 << 16));    // tile_dim0
      g1[4] = (int)(tl1);                          // tile_dim1 (tile_dim2 = 0)
      g1[5] = (int)st0;                            // dim0_stride[31:0]
      g1[6] = 0;                                   // stride[47:32], dim1_stride lo
      g1[7] = 0;
      i32x4 z4 = {0, 0, 0, 0};
      i32x8 z8 = {0, 0, 0, 0, 0, 0, 0, 0};
      __builtin_amdgcn_tensor_load_to_lds(g0, g1, z4, z4, z8, 0);
      __builtin_amdgcn_s_wait_tensorcnt((short)0);
    }
  } else
#endif
  {
    for (int u = tid; u < 6 * 260 * 4; u += 256) {
      int ck = u & 3;
      int c  = (u >> 2) % 260;
      int r  = u / (260 * 4);
      int gy = gy0 + r;
      int gx = gx0 + c;
      u16x8 pk;
      if (gy >= 0 && gy < HH && gx >= 0 && gx < WW)
        pk = *(const u16x8*)(h + (((size_t)(n * HH + gy) * WW + gx) * CO + ck * 8));
      else
        pk = (u16x8)0;
      *(u16x8*)(smem + ((r * 260 + c) * 32 + ck * 8)) = pk;
    }
  }
  __syncthreads();

  const int wave = tid >> 5;
  const int lane = tid & 31;
  const int m    = lane & 15;
  const int hi   = lane >> 4;
  const int run0 = hi * 8;
  const int run1 = 16 + hi * 8;
  const int ox   = ob0 + wave * 16 + m;

  v8f acc0 = {0,0,0,0,0,0,0,0};
  v8f acc1 = {0,0,0,0,0,0,0,0};

  // Main path: 6x6 effective taps, each = 2 WMMAs (N-tiles 0..15 / 16..31).
  #pragma unroll
  for (int u = 0; u < 6; ++u) {
    #pragma unroll
    for (int v = 0; v < 6; ++v) {
      const u16* ap = smem + ((u * 260 + (2 * (wave * 16 + m) + v)) * 32);
      v16bf a = cat16(*(const u16x8*)(ap + run0), *(const u16x8*)(ap + run1));

      const u16* bp = weff + ((u * 6 + v) * 1024 + m * 32);
      v16bf bv0 = cat16(*(const u16x8*)(bp + run0), *(const u16x8*)(bp + run1));
      v16bf bv1 = cat16(*(const u16x8*)(bp + 512 + run0),
                        *(const u16x8*)(bp + 512 + run1));

      acc0 = __builtin_amdgcn_wmma_f32_16x16x32_bf16(false, a, false, bv0,
                                                     (short)0, acc0, false, false);
      acc1 = __builtin_amdgcn_wmma_f32_16x16x32_bf16(false, a, false, bv1,
                                                     (short)0, acc1, false, false);
    }
  }

  // Residual path: xb[m, c] = sum_{i,j} F[i,j] * x[2oy+i-1, 2ox+j-1, c]
  float xs[16];
  #pragma unroll
  for (int e = 0; e < 16; ++e) xs[e] = 0.0f;
  const float K4[4] = {0.125f, 0.375f, 0.375f, 0.125f};
  #pragma unroll
  for (int i = 0; i < 4; ++i) {
    const int gy = 2 * oy + i - 1;
    if (gy < 0 || gy >= HH) continue;
    #pragma unroll
    for (int j = 0; j < 4; ++j) {
      const int gx = 2 * ox + j - 1;
      if (gx < 0 || gx >= WW) continue;
      const float cf = K4[i] * K4[j];
      const float* src = x + (((size_t)(n * HH + gy) * WW + gx) * CI);
      f32x4 p0 = *(const f32x4*)(src + run0);
      f32x4 p1 = *(const f32x4*)(src + run0 + 4);
      f32x4 p2 = *(const f32x4*)(src + run1);
      f32x4 p3 = *(const f32x4*)(src + run1 + 4);
      xs[0]  += cf * p0.x; xs[1]  += cf * p0.y; xs[2]  += cf * p0.z; xs[3]  += cf * p0.w;
      xs[4]  += cf * p1.x; xs[5]  += cf * p1.y; xs[6]  += cf * p1.z; xs[7]  += cf * p1.w;
      xs[8]  += cf * p2.x; xs[9]  += cf * p2.y; xs[10] += cf * p2.z; xs[11] += cf * p2.w;
      xs[12] += cf * p3.x; xs[13] += cf * p3.y; xs[14] += cf * p3.z; xs[15] += cf * p3.w;
    }
  }
  u16x16 arr;
  #pragma unroll
  for (int e = 0; e < 16; ++e) arr[e] = f2bf(xs[e]);
  v16bf ar = __builtin_bit_cast(v16bf, arr);

  const u16* rp = wresb + m * 32;
  v16bf rv0 = cat16(*(const u16x8*)(rp + run0), *(const u16x8*)(rp + run1));
  v16bf rv1 = cat16(*(const u16x8*)(rp + 512 + run0),
                    *(const u16x8*)(rp + 512 + run1));
  v8f r0 = {0,0,0,0,0,0,0,0};
  v8f r1 = {0,0,0,0,0,0,0,0};
  r0 = __builtin_amdgcn_wmma_f32_16x16x32_bf16(false, ar, false, rv0,
                                               (short)0, r0, false, false);
  r1 = __builtin_amdgcn_wmma_f32_16x16x32_bf16(false, ar, false, rv1,
                                               (short)0, r1, false, false);

  const float bias0 = bias[m];
  const float bias1 = bias[m + 16];
  const float RS = 0.7071067811865476f;
  float* op = out + (((size_t)(n * HO + oy) * WO + (ob0 + wave * 16 + hi * 8)) * CO + m);
  #pragma unroll
  for (int g = 0; g < 8; ++g) {
    op[(size_t)g * CO]      = (lrelu_gain(acc0[g] + bias0) + r0[g]) * RS;
    op[(size_t)g * CO + 16] = (lrelu_gain(acc1[g] + bias1) + r1[g]) * RS;
  }
}

// ---------------------------------------------------------------------------
extern "C" void kernel_launch(void* const* d_in, const int* in_sizes, int n_in,
                              void* d_out, int out_size, void* d_ws, size_t ws_size,
                              hipStream_t stream) {
  (void)in_sizes; (void)n_in; (void)out_size; (void)ws_size;
  const float* x    = (const float*)d_in[0];
  const float* w0   = (const float*)d_in[1];
  const float* b0   = (const float*)d_in[2];
  const float* w1   = (const float*)d_in[3];
  const float* b1   = (const float*)d_in[4];
  const float* wres = (const float*)d_in[5];
  float* out = (float*)d_out;

  // Workspace layout (bf16 elements): h, then weight tables.
  u16* h     = (u16*)d_ws;                         // 8*512*512*32
  u16* w0b   = h + (size_t)NB * HH * WW * CO;      // 9*32*32
  u16* weff  = w0b + 9 * 32 * 32;                  // 36*32*32
  u16* wresb = weff + 36 * 32 * 32;                // 32*32

  prep_weights<<<64, 256, 0, stream>>>(w0, w1, wres, w0b, weff, wresb);

  const int grid0 = NB * HH * (WW / 128);          // 16384
  conv0_lrelu<<<grid0, 256, 3 * 130 * 32 * (int)sizeof(u16), stream>>>(
      x, b0, w0b, h);

  const int grid1 = NB * HO * (WO / 128);          // 4096
  down_conv_res<<<grid1, 256, 6 * 260 * 32 * (int)sizeof(u16), stream>>>(
      h, x, b1, weff, wresb, out);
}
